// ImageEncoderV2_2370821948085
// MI455X (gfx1250) — matrix operand order repaired and checked by reading, not compiled, not run on hardware
//
#include <hip/hip_runtime.h>

// ---- problem constants (from reference setup_inputs) ----
#define BB   16
#define CC   512
#define HWp  256      // h*w = 16*16
#define EE   128
#define NNo  32
#define HH   512
#define WWd  512
#define hh_  16
#define ww_  16

typedef __bf16 bf16_t;
typedef bf16_t v16bf __attribute__((ext_vector_type(16)));
typedef bf16_t v8bf  __attribute__((ext_vector_type(8)));
typedef float  v8f   __attribute__((ext_vector_type(8)));

// ---------------------------------------------------------------------------
// WMMA operand loaders (contiguous k order, same for A and B; wave32)
// A 16x32 bf16, row-major source (rows = M, K contiguous):
//   lane m = lane&15 ; half = lane>>4 ; elems 0..15 -> K = k0+16*half+j.
__device__ __forceinline__ v16bf load_a16x32(const bf16_t* row, int k0, int half) {
  v8bf c1 = *(const v8bf*)(row + k0 + 16 * half);
  v8bf c2 = *(const v8bf*)(row + k0 + 16 * half + 8);
  v16bf a;
#pragma unroll
  for (int i = 0; i < 8; ++i) { a[i] = c1[i]; a[i + 8] = c2[i]; }
  return a;
}
// B 32x16 bf16 where source row holds K contiguous for a fixed column:
//   lane col n = lane&15, K = k0 + 16*half + j  (16 contiguous elems).
__device__ __forceinline__ v16bf load_b32x16(const bf16_t* colrow, int k0, int half) {
  return *(const v16bf*)(colrow + k0 + 16 * half);
}
// Same as load_b32x16 but converting from contiguous f32 on the fly.
__device__ __forceinline__ v16bf load_b32x16_f32(const float* p) {
  v16bf r;
#pragma unroll
  for (int i = 0; i < 16; ++i) r[i] = (bf16_t)p[i];
  return r;
}

__device__ __forceinline__ v8f wmma_bf16(v16bf a, v16bf b, v8f c) {
  return __builtin_amdgcn_wmma_f32_16x16x32_bf16(false, a, false, b, (short)0, c,
                                                 false, false);
}

// ---------------------------------------------------------------------------
// Prep: fused BN scale/bias  (scale = gamma/sqrt(var+eps),
//                             bias  = conv_b*scale + beta - mean*scale)
__global__ void k_prep_bn(const float* __restrict__ conv_b,
                          const float* __restrict__ gamma,
                          const float* __restrict__ beta,
                          const float* __restrict__ mean,
                          const float* __restrict__ var,
                          float* __restrict__ scale_o, float* __restrict__ bias_o) {
  int e = threadIdx.x;
  if (e < EE) {
    float s = gamma[e] / sqrtf(var[e] + 1e-5f);
    scale_o[e] = s;
    bias_o[e] = conv_b[e] * s + beta[e] - mean[e] * s;
  }
}

// conv_w [E,C] f32 -> bf16 same layout
__global__ void k_cvt_convw(const float* __restrict__ in, bf16_t* __restrict__ o) {
  int i = blockIdx.x * 256 + threadIdx.x;
  if (i < EE * CC) o[i] = (bf16_t)in[i];
}

// features [b][c][hw] f32 -> feat_t bf16 [b][hw][c]
__global__ void k_cvt_feat(const float* __restrict__ in, bf16_t* __restrict__ o) {
  int i = blockIdx.x * 256 + threadIdx.x;
  if (i >= BB * CC * HWp) return;
  int hw = i % HWp;
  int c = (i / HWp) % CC;
  int b = i / (CC * HWp);
  o[(b * HWp + hw) * CC + c] = (bf16_t)in[i];
}

// ---------------------------------------------------------------------------
// Sparse bilinear resize: 2 taps per axis (matches _resize_mat exactly).
// One block per (b,n); thread (o,p); also reduces mask_sum -> 1/(sum+eps).
__global__ void k_resize(const float* __restrict__ masks,
                         bf16_t* __restrict__ mr, float* __restrict__ inv_sum) {
  __shared__ float red[256];
  int bn = blockIdx.x;            // b*NNo + n
  int t = threadIdx.x;            // 0..255
  int o = t >> 4, p = t & 15;
  const float sH = (float)HH / (float)hh_;
  const float sW = (float)WWd / (float)ww_;
  float sy = fmaxf((o + 0.5f) * sH - 0.5f, 0.0f);
  float sx = fmaxf((p + 0.5f) * sW - 0.5f, 0.0f);
  int i0 = (int)floorf(sy); float fy = sy - (float)i0; int i1 = min(i0 + 1, HH - 1);
  int j0 = (int)floorf(sx); float fx = sx - (float)j0; int j1 = min(j0 + 1, WWd - 1);
  const float* m = masks + (size_t)bn * HH * WWd;
  float v00 = m[i0 * WWd + j0], v01 = m[i0 * WWd + j1];
  float v10 = m[i1 * WWd + j0], v11 = m[i1 * WWd + j1];
  float val = (1.f - fy) * ((1.f - fx) * v00 + fx * v01)
            + fy * ((1.f - fx) * v10 + fx * v11);
  mr[(size_t)bn * HWp + t] = (bf16_t)val;
  red[t] = val;
  __syncthreads();
#pragma unroll
  for (int s = 128; s > 0; s >>= 1) {
    if (t < s) red[t] += red[t + s];
    __syncthreads();
  }
  if (t == 0) inv_sum[bn] = 1.0f / (red[0] + 1e-8f);
}

// ---------------------------------------------------------------------------
// Projection GEMM: per batch  D[e,hw] = conv_w[e,:] . feat[b,:,hw]
// then BN-affine + ReLU -> proj_bf16 [b][e][hw].
// One wave per 16x16 tile over K=512. 2048 tiles / 8 waves = 256 blocks.
__global__ void k_proj(const bf16_t* __restrict__ cw,      // [E][C]
                       const bf16_t* __restrict__ feat_t,  // [B][HW][C]
                       const float* __restrict__ scale, const float* __restrict__ bias,
                       bf16_t* __restrict__ proj) {        // [B][E][HW]
  int wg = blockIdx.x * 8 + (threadIdx.x >> 5);
  int lane = threadIdx.x & 31, l15 = lane & 15, half = lane >> 4;
  int b = wg >> 7;                 // /128
  int rem = wg & 127;
  int e0 = (rem >> 4) * 16;        // m-tile
  int hw0 = (rem & 15) * 16;       // n-tile
  const bf16_t* arow = cw + (size_t)(e0 + l15) * CC;
  const bf16_t* brow = feat_t + ((size_t)b * HWp + hw0 + l15) * CC;
  v8f acc = {};
#pragma unroll 4
  for (int k0 = 0; k0 < CC; k0 += 32)
    acc = wmma_bf16(load_a16x32(arow, k0, half), load_b32x16(brow, k0, half), acc);
#pragma unroll
  for (int v = 0; v < 8; ++v) {
    int e = e0 + v + 8 * half;
    float val = acc[v] * scale[e] + bias[e];
    val = fmaxf(val, 0.0f);
    proj[((size_t)b * EE + e) * HWp + hw0 + l15] = (bf16_t)val;
  }
}

// ---------------------------------------------------------------------------
// Pooling GEMM: per batch  obj[n,e] = (proj[e,:] . mr[n,:]) * inv_sum[b,n]
// D tiles [e-tile(8) x n-tile(2)] per batch, K=256. 256 tiles -> 32 blocks.
__global__ void k_pool(const bf16_t* __restrict__ proj,  // [B][E][HW]
                       const bf16_t* __restrict__ mr,    // [B][N][HW]
                       const float* __restrict__ inv_sum,
                       bf16_t* __restrict__ obj) {       // [B*N][E]
  int wg = blockIdx.x * 8 + (threadIdx.x >> 5);
  int lane = threadIdx.x & 31, l15 = lane & 15, half = lane >> 4;
  int b = wg >> 4;
  int rem = wg & 15;
  int e0 = (rem >> 1) * 16;
  int n0 = (rem & 1) * 16;
  const bf16_t* arow = proj + ((size_t)b * EE + e0 + l15) * HWp;
  const bf16_t* brow = mr + ((size_t)b * NNo + n0 + l15) * HWp;
  v8f acc = {};
#pragma unroll 4
  for (int k0 = 0; k0 < HWp; k0 += 32)
    acc = wmma_bf16(load_a16x32(arow, k0, half), load_b32x16(brow, k0, half), acc);
  int n = n0 + l15;                       // D col = n
  float inv = inv_sum[b * NNo + n];
#pragma unroll
  for (int v = 0; v < 8; ++v) {
    int e = e0 + v + 8 * half;            // D row = e
    obj[((size_t)b * NNo + n) * EE + e] = (bf16_t)(acc[v] * inv);
  }
}

// ---------------------------------------------------------------------------
// Fused 2-layer MLP: rows = B*N = 512, E=128, K=128.
// Block = 16 rows x all 128 cols; wave w owns e-tile w. hdn staged in LDS.
__global__ void k_mlp(const bf16_t* __restrict__ obj,  // [512][128] bf16
                      const float* __restrict__ w1, const float* __restrict__ b1,
                      const float* __restrict__ w2, const float* __restrict__ b2,
                      float* __restrict__ out) {       // [512][128] f32
  __shared__ bf16_t hdn[16 * EE];
  int warp = threadIdx.x >> 5;
  int lane = threadIdx.x & 31, l15 = lane & 15, half = lane >> 4;
  int r0 = blockIdx.x * 16;
  int e0 = warp * 16;
  // layer 1: y = relu(obj @ w1.T + b1)
  {
    const bf16_t* arow = obj + (size_t)(r0 + l15) * EE;
    const float* bcol = w1 + (size_t)(e0 + l15) * EE;   // w1[e, k], k contiguous
    v8f acc = {};
#pragma unroll
    for (int k0 = 0; k0 < EE; k0 += 32)
      acc = wmma_bf16(load_a16x32(arow, k0, half),
                      load_b32x16_f32(bcol + k0 + 16 * half), acc);
    float bv = b1[e0 + l15];
#pragma unroll
    for (int v = 0; v < 8; ++v) {
      int r_in = v + 8 * half;                          // D row = r (A rows)
      hdn[r_in * EE + e0 + l15] = (bf16_t)fmaxf(acc[v] + bv, 0.0f);
    }
  }
  __syncthreads();
  // layer 2: out = hdn @ w2.T + b2
  {
    const bf16_t* arow = hdn + (size_t)l15 * EE;        // LDS, 16B chunks
    const float* bcol = w2 + (size_t)(e0 + l15) * EE;
    v8f acc = {};
#pragma unroll
    for (int k0 = 0; k0 < EE; k0 += 32)
      acc = wmma_bf16(load_a16x32(arow, k0, half),
                      load_b32x16_f32(bcol + k0 + 16 * half), acc);
    float bv = b2[e0 + l15];
#pragma unroll
    for (int v = 0; v < 8; ++v) {
      int r = r0 + v + 8 * half;
      out[(size_t)r * EE + e0 + l15] = acc[v] + bv;
    }
  }
}

// ---------------------------------------------------------------------------
extern "C" void kernel_launch(void* const* d_in, const int* in_sizes, int n_in,
                              void* d_out, int out_size, void* d_ws, size_t ws_size,
                              hipStream_t stream) {
  const float* features = (const float*)d_in[0];
  const float* masks    = (const float*)d_in[1];
  const float* conv_w   = (const float*)d_in[2];
  const float* conv_b   = (const float*)d_in[3];
  const float* bn_gamma = (const float*)d_in[4];
  const float* bn_beta  = (const float*)d_in[5];
  const float* bn_mean  = (const float*)d_in[6];
  const float* bn_var   = (const float*)d_in[7];
  const float* w1 = (const float*)d_in[8];
  const float* b1 = (const float*)d_in[9];
  const float* w2 = (const float*)d_in[10];
  const float* b2 = (const float*)d_in[11];
  float* out = (float*)d_out;

  char* ws = (char*)d_ws;
  size_t off = 0;
  auto take = [&](size_t bytes) -> void* {
    void* p = ws + off;
    off = (off + bytes + 255) & ~(size_t)255;
    return p;
  };
  bf16_t* feat_t  = (bf16_t*)take((size_t)BB * HWp * CC * 2);   // 4 MB
  bf16_t* cw_bf   = (bf16_t*)take((size_t)EE * CC * 2);         // 128 KB
  bf16_t* mr      = (bf16_t*)take((size_t)BB * NNo * HWp * 2);  // 256 KB
  float*  inv_sum = (float*) take((size_t)BB * NNo * 4);
  bf16_t* proj    = (bf16_t*)take((size_t)BB * EE * HWp * 2);   // 1 MB
  bf16_t* obj     = (bf16_t*)take((size_t)BB * NNo * EE * 2);   // 128 KB
  float*  scale   = (float*) take(EE * 4);
  float*  bias    = (float*) take(EE * 4);

  k_prep_bn<<<1, 128, 0, stream>>>(conv_b, bn_gamma, bn_beta, bn_mean, bn_var,
                                   scale, bias);
  k_cvt_convw<<<(EE * CC + 255) / 256, 256, 0, stream>>>(conv_w, cw_bf);
  k_cvt_feat<<<(BB * CC * HWp + 255) / 256, 256, 0, stream>>>(features, feat_t);
  k_resize<<<BB * NNo, 256, 0, stream>>>(masks, mr, inv_sum);
  k_proj<<<256, 256, 0, stream>>>(cw_bf, feat_t, scale, bias, proj);
  k_pool<<<32, 256, 0, stream>>>(proj, mr, inv_sum, obj);
  k_mlp<<<(BB * NNo) / 16, 256, 0, stream>>>(obj, w1, b1, w2, b2, out);
}
